// MACE_82532091560022
// MI455X (gfx1250) — matrix operand order repaired and checked by reading, not compiled, not run on hardware
//
#include <hip/hip_runtime.h>
#include <math.h>

// ---------------- problem constants (match reference) ----------------
#define N_NODES   10000
#define N_EDGES   160000
#define N_GRAPHS  16
#define EMB       16
#define MLP_DIM   256
#define OUT_CH    1195
#define NUM_AA    26
#define NUM_BB    6
#define R_MAX_INV 0.1f

#define SQ3F      1.7320508075688772f
#define INV_SQ3F  0.5773502691896258f
#define INV_SQ2F  0.7071067811865475f
#define PIF       3.14159265358979323846f

// WMMA operand vector types (CDNA5, wave32)
typedef __attribute__((ext_vector_type(16))) __bf16 v16bf;
typedef __attribute__((ext_vector_type(8)))  float  v8f;

// round-to-nearest-even f32 -> bf16 (bit manipulation; no dependence on cvt support)
__device__ __forceinline__ __bf16 f2bf(float x) {
    unsigned u = __builtin_bit_cast(unsigned, x);
    unsigned r = (u + 0x7FFFu + ((u >> 16) & 1u)) >> 16;
    return __builtin_bit_cast(__bf16, (unsigned short)r);
}

// ---------------------------------------------------------------------
// utility: zero fill
__global__ void k_zero(float* p, int n) {
    int i = blockIdx.x * blockDim.x + threadIdx.x;
    if (i < n) p[i] = 0.f;
}

// ---------------------------------------------------------------------
// node embedding: s = [onehot(z) ++ bb] @ W_emb + b
__global__ void k_embed(const int* __restrict__ z, const float* __restrict__ bb,
                        const float* __restrict__ Wemb, const float* __restrict__ bemb,
                        float* __restrict__ sN) {
    int n = blockIdx.x * blockDim.x + threadIdx.x;
    if (n >= N_NODES) return;
    int zi = z[n];
    float bl[NUM_BB];
#pragma unroll
    for (int j = 0; j < NUM_BB; ++j) bl[j] = bb[n * NUM_BB + j];
#pragma unroll
    for (int o = 0; o < EMB; ++o) {
        float a = Wemb[zi * EMB + o] + bemb[o];
#pragma unroll
        for (int j = 0; j < NUM_BB; ++j) a += bl[j] * Wemb[(NUM_AA + j) * EMB + o];
        sN[n * EMB + o] = a;
    }
}

// ---------------------------------------------------------------------
// edge geometry: y1 (l=1 SH, e3nn y,z,x order) and Bessel*envelope feats
__global__ void k_edge_geom(const int* __restrict__ ei, const float* __restrict__ pos,
                            float* __restrict__ y1g, float* __restrict__ fts) {
    int e = blockIdx.x * blockDim.x + threadIdx.x;
    if (e >= N_EDGES) return;
    int si = ei[e], di = ei[N_EDGES + e];
    float vx = pos[si * 3 + 0] - pos[di * 3 + 0];
    float vy = pos[si * 3 + 1] - pos[di * 3 + 1];
    float vz = pos[si * 3 + 2] - pos[di * 3 + 2];
    float r  = sqrtf(vx * vx + vy * vy + vz * vz);
    float inv = 1.f / r;
    y1g[e * 3 + 0] = SQ3F * vy * inv;   // (y,z,x) order
    y1g[e * 3 + 1] = SQ3F * vz * inv;
    y1g[e * 3 + 2] = SQ3F * vx * inv;
    float u = r * R_MAX_INV;
    float env = 0.f;
    if (u < 1.f) {
        float u2 = u * u;
        float u5 = u2 * u2 * u;
        env = 1.f + u5 * (-21.f + u * (35.f - 15.f * u));   // p=5 polynomial cutoff
    }
    float c = 0.4472135954999579f * inv * env;              // sqrt(2/R_MAX)=sqrt(0.2)
#pragma unroll
    for (int j = 0; j < 8; ++j)
        fts[(size_t)e * 8 + j] = c * sinf((float)(j + 1) * PIF * u);
}

// ---------------------------------------------------------------------
// pack mlp2 weights (f32, [256, ncols]) into bf16 in this kernel's contiguous-k WMMA B-operand
// lane layout:  out[((ct*8 + t)*32 + lane)*16 + e]
//   row = 32*t + (lane>=16 ? 16 : 0) + e ,  col = ct*16 + (lane&15)
// -> each lane's B fragment for one k-step is one contiguous 32B load.
__global__ void k_pack_w2(const float* __restrict__ W2, __bf16* __restrict__ outp,
                          int ncols, int nelem) {
    int m = blockIdx.x * blockDim.x + threadIdx.x;
    if (m >= nelem) return;
    int e  = m & 15;
    int ln = (m >> 4) & 31;
    int t  = (m >> 9) & 7;
    int ct = m >> 12;
    int row = 32 * t + ((ln >= 16) ? 16 : 0) + e;
    int col = ct * 16 + (ln & 15);
    outp[m] = f2bf(W2[(size_t)row * ncols + col]);
}

// ---------------------------------------------------------------------
// THE fused edge kernel. One wave = one 16-edge tile (WMMA M dimension).
//  1) hid[16,256] = relu(feats @ mlp1_W + b)  -- computed per lane in f32
//     directly in the A-operand lane layout, converted to bf16 fragments.
//  2) for each of P*16 column chunks of mlp2_W: 8 chained
//     v_wmma_f32_16x16x32_bf16 over K=256 give w-slice[16e,16o]; it is
//     immediately contracted with per-edge s/dots/v/cross (LDS broadcasts)
//     into the message accumulators -- the [E,P,16,16] tensor is never stored.
//  3) scale + scatter-add to node accumulators with global f32 atomics.
template <bool FIRST>
__global__ __launch_bounds__(128) void k_edge_msg(
    const int* __restrict__ eSrc, const int* __restrict__ eDst,
    const float* __restrict__ y1g, const float* __restrict__ fts,
    const float* __restrict__ sN, const float* __restrict__ vN,
    const float* __restrict__ W1, const float* __restrict__ b1,
    const __bf16* __restrict__ w2, const float* __restrict__ b2,
    float* __restrict__ hs, float* __restrict__ hv, float a0, float a1) {

    __shared__ __align__(16) float W1s[8 * 256];
    __shared__ __align__(16) float b1s[256];
    __shared__ float sT[4][256];     // gathered s[dst]  : [edge][i]
    __shared__ float vT[4][768];     // gathered v[dst]  : [edge][i][c]
    __shared__ float y1T[4][64];     // y1 per edge      : [edge][c] (pad 4)

    const int tid = threadIdx.x;
    const int wv  = tid >> 5;
    const int lane = tid & 31;
    const int e0 = (blockIdx.x * 4 + wv) * 16;

    for (int i = tid; i < 8 * 256; i += 128) W1s[i] = W1[i];
    for (int i = tid; i < 256; i += 128) b1s[i] = b1[i];

    for (int idx = lane; idx < 256; idx += 32) {
        int e = idx >> 4, i = idx & 15;
        int d = eDst[e0 + e];
        sT[wv][idx] = sN[d * 16 + i];
        if constexpr (!FIRST) {
            const float* vp = &vN[(size_t)(d * 16 + i) * 3];
            vT[wv][idx * 3 + 0] = vp[0];
            vT[wv][idx * 3 + 1] = vp[1];
            vT[wv][idx * 3 + 2] = vp[2];
        }
    }
    if (lane < 16) {
        y1T[wv][lane * 4 + 0] = y1g[(size_t)(e0 + lane) * 3 + 0];
        y1T[wv][lane * 4 + 1] = y1g[(size_t)(e0 + lane) * 3 + 1];
        y1T[wv][lane * 4 + 2] = y1g[(size_t)(e0 + lane) * 3 + 2];
        y1T[wv][lane * 4 + 3] = 0.f;
    }
    __syncthreads();

    const int er = lane & 15;       // edge row (A-operand M index)
    const int hi = lane >> 4;       // lane-half selects K sub-range

    float f[8];
    {
        const float* fp = &fts[(size_t)(e0 + er) * 8];
#pragma unroll
        for (int j = 0; j < 8; ++j) f[j] = fp[j];
    }

    // hid in A-operand bf16 fragments: lane<16 holds K = 32t+{0..15},
    // lane>=16 holds K = 32t+{16..31} (contiguous order, matching packed B).
    v16bf hidA[8];
#pragma unroll
    for (int t = 0; t < 8; ++t) {
#pragma unroll
        for (int hsec = 0; hsec < 2; ++hsec) {
            const int h0 = 32 * t + hi * 16 + hsec * 8;
            float acc[8];
            {
                const float4* b4 = (const float4*)&b1s[h0];
                float4 x = b4[0], y = b4[1];
                acc[0] = x.x; acc[1] = x.y; acc[2] = x.z; acc[3] = x.w;
                acc[4] = y.x; acc[5] = y.y; acc[6] = y.z; acc[7] = y.w;
            }
#pragma unroll
            for (int j = 0; j < 8; ++j) {
                const float4* w4 = (const float4*)&W1s[j * 256 + h0];
                float4 x = w4[0], y = w4[1];
                float fj = f[j];
                acc[0] += fj * x.x; acc[1] += fj * x.y;
                acc[2] += fj * x.z; acc[3] += fj * x.w;
                acc[4] += fj * y.x; acc[5] += fj * y.y;
                acc[6] += fj * y.z; acc[7] += fj * y.w;
            }
#pragma unroll
            for (int q = 0; q < 8; ++q)
                hidA[t][hsec * 8 + q] = f2bf(fmaxf(acc[q], 0.f));
        }
    }

    // message accumulators (C-layout: element r <-> edge r + hi*8, o = lane&15)
    float m_s[8], tv[8], tsy[8], mv0[8], mv1[8], mv2[8];
#pragma unroll
    for (int r = 0; r < 8; ++r) {
        m_s[r] = 0.f; tv[r] = 0.f; tsy[r] = 0.f;
        mv0[r] = 0.f; mv1[r] = 0.f; mv2[r] = 0.f;
    }

    const int P = FIRST ? 2 : 5;
    const int o = lane & 15;

    for (int p = 0; p < P; ++p) {
        for (int i = 0; i < 16; ++i) {
            const int ct = p * 16 + i;
            const float bbias = b2[ct * 16 + o];
            v8f acc;
#pragma unroll
            for (int k = 0; k < 8; ++k) acc[k] = bbias;

            const __bf16* wp = w2 + (size_t)ct * (8 * 32 * 16);
            __builtin_prefetch(wp + 8 * 32 * 16, 0, 1);   // next chunk -> global_prefetch
#pragma unroll
            for (int t = 0; t < 8; ++t) {
                v16bf B = *reinterpret_cast<const v16bf*>(wp + (size_t)(t * 32 + lane) * 16);
                acc = __builtin_amdgcn_wmma_f32_16x16x32_bf16(
                    false, hidA[t], false, B, (short)0, acc, false, false);
            }

            // contract w-slice with path-specific per-edge left operand
#pragma unroll
            for (int r = 0; r < 8; ++r) {
                const int eIdx = r + hi * 8;
                const float wvv = acc[r];
                if constexpr (FIRST) {
                    const float u = sT[wv][eIdx * 16 + i];
                    if (p == 0) m_s[r] += u * wvv;
                    else        tv[r]  += u * wvv;
                } else {
                    switch (p) {
                    case 0: m_s[r] += sT[wv][eIdx * 16 + i] * wvv; break;
                    case 1: {
                        const float* vvp = &vT[wv][(eIdx * 16 + i) * 3];
                        const float* yy  = &y1T[wv][eIdx * 4];
                        float d = (vvp[0] * yy[0] + vvp[1] * yy[1] + vvp[2] * yy[2]) * INV_SQ3F;
                        m_s[r] += d * wvv; break;
                    }
                    case 2: tsy[r] += sT[wv][eIdx * 16 + i] * wvv; break;
                    case 3: {
                        const float* vvp = &vT[wv][(eIdx * 16 + i) * 3];
                        mv0[r] += vvp[0] * wvv; mv1[r] += vvp[1] * wvv; mv2[r] += vvp[2] * wvv;
                        break;
                    }
                    case 4: {
                        const float* vvp = &vT[wv][(eIdx * 16 + i) * 3];
                        const float* yy  = &y1T[wv][eIdx * 4];
                        float cx = (vvp[1] * yy[2] - vvp[2] * yy[1]) * INV_SQ2F;
                        float cy = (vvp[2] * yy[0] - vvp[0] * yy[2]) * INV_SQ2F;
                        float cz = (vvp[0] * yy[1] - vvp[1] * yy[0]) * INV_SQ2F;
                        mv0[r] += cx * wvv; mv1[r] += cy * wvv; mv2[r] += cz * wvv;
                        break;
                    }
                    }
                }
            }
        }
    }

    // scale + scatter-add to nodes (segment over edge src)
#pragma unroll
    for (int r = 0; r < 8; ++r) {
        const int eIdx = r + hi * 8;
        const int sn = eSrc[e0 + eIdx];
        const float* yy = &y1T[wv][eIdx * 4];
        if constexpr (FIRST) {
            atomicAdd(&hs[sn * 16 + o], a0 * m_s[r]);
            const float t0 = a0 * tv[r];
            atomicAdd(&hv[(size_t)(sn * 16 + o) * 3 + 0], t0 * yy[0]);
            atomicAdd(&hv[(size_t)(sn * 16 + o) * 3 + 1], t0 * yy[1]);
            atomicAdd(&hv[(size_t)(sn * 16 + o) * 3 + 2], t0 * yy[2]);
        } else {
            atomicAdd(&hs[sn * 16 + o], a0 * m_s[r]);
            atomicAdd(&hv[(size_t)(sn * 16 + o) * 3 + 0], a1 * (mv0[r] + tsy[r] * yy[0]));
            atomicAdd(&hv[(size_t)(sn * 16 + o) * 3 + 1], a1 * (mv1[r] + tsy[r] * yy[1]));
            atomicAdd(&hv[(size_t)(sn * 16 + o) * 3 + 2], a1 * (mv2[r] + tsy[r] * yy[2]));
        }
    }
}

// ---------------------------------------------------------------------
// symmetric contraction (corr=3, lmax=1) + linear + residual, per node
__global__ void k_node_update(const float* __restrict__ hs, const float* __restrict__ hv,
                              const float* __restrict__ w0, const float* __restrict__ w1,
                              const float* __restrict__ linWs, const float* __restrict__ linWv,
                              float* __restrict__ sN, float* __restrict__ vN) {
    int n = blockIdx.x * blockDim.x + threadIdx.x;
    if (n >= N_NODES) return;
    float hsv[16], hvv[48], p0[16], p1f[16];
#pragma unroll
    for (int k = 0; k < 16; ++k) hsv[k] = hs[n * 16 + k];
#pragma unroll
    for (int k = 0; k < 48; ++k) hvv[k] = hv[(size_t)n * 48 + k];
#pragma unroll
    for (int k = 0; k < 16; ++k) {
        float a = hsv[k], a2 = a * a;
        float vvn = hvv[k * 3] * hvv[k * 3] + hvv[k * 3 + 1] * hvv[k * 3 + 1] +
                    hvv[k * 3 + 2] * hvv[k * 3 + 2];
        p0[k]  = w0[k] * a + w0[16 + k] * a2 + w0[32 + k] * vvn +
                 w0[48 + k] * a2 * a + w0[64 + k] * a * vvn;
        p1f[k] = w1[k] + w1[16 + k] * a + w1[32 + k] * a2 + w1[48 + k] * vvn;
    }
#pragma unroll
    for (int oo = 0; oo < 16; ++oo) {
        float acc = 0.f;
#pragma unroll
        for (int k = 0; k < 16; ++k) acc += p0[k] * linWs[k * 16 + oo];
        sN[n * 16 + oo] += 0.25f * acc;     // 1/sqrt(EMB)
    }
#pragma unroll
    for (int oo = 0; oo < 16; ++oo) {
        float a0c = 0.f, a1c = 0.f, a2c = 0.f;
#pragma unroll
        for (int k = 0; k < 16; ++k) {
            float pw = p1f[k] * linWv[k * 16 + oo];
            a0c += pw * hvv[k * 3 + 0];
            a1c += pw * hvv[k * 3 + 1];
            a2c += pw * hvv[k * 3 + 2];
        }
        vN[(size_t)(n * 16 + oo) * 3 + 0] += 0.25f * a0c;
        vN[(size_t)(n * 16 + oo) * 3 + 1] += 0.25f * a1c;
        vN[(size_t)(n * 16 + oo) * 3 + 2] += 0.25f * a2c;
    }
}

// ---------------------------------------------------------------------
// sum-pool [s ++ v] per graph
__global__ void k_pool(const float* __restrict__ sN, const float* __restrict__ vN,
                       const int* __restrict__ batch, float* __restrict__ g) {
    int n = blockIdx.x * blockDim.x + threadIdx.x;
    if (n >= N_NODES) return;
    int b = batch[n];
#pragma unroll
    for (int oo = 0; oo < 16; ++oo) atomicAdd(&g[b * 64 + oo], sN[n * 16 + oo]);
#pragma unroll
    for (int k = 0; k < 48; ++k) atomicAdd(&g[b * 64 + 16 + k], vN[(size_t)n * 48 + k]);
}

// ---------------------------------------------------------------------
// readout MLP (tiny): relu(g @ out1_W + b) @ out2_W + b
__global__ __launch_bounds__(256) void k_readout(const float* __restrict__ g,
                                                 const float* __restrict__ W1o,
                                                 const float* __restrict__ b1o,
                                                 const float* __restrict__ W2o,
                                                 const float* __restrict__ b2o,
                                                 float* __restrict__ out) {
    __shared__ float gS[16 * 64];
    __shared__ float o1S[16 * 256];
    int tid = threadIdx.x;
    for (int i = tid; i < 1024; i += 256) gS[i] = g[i];
    __syncthreads();
    for (int i = tid; i < 4096; i += 256) {
        int r = i >> 8, c = i & 255;
        float a = b1o[c];
        for (int k = 0; k < 64; ++k) a += gS[r * 64 + k] * W1o[k * 256 + c];
        o1S[i] = fmaxf(a, 0.f);
    }
    __syncthreads();
    for (int i = tid; i < 16 * OUT_CH; i += 256) {
        int r = i / OUT_CH, c = i % OUT_CH;
        float a = b2o[c];
        for (int k = 0; k < 256; ++k) a += o1S[r * 256 + k] * W2o[k * OUT_CH + c];
        out[i] = a;
    }
}

// ---------------------------------------------------------------------
extern "C" void kernel_launch(void* const* d_in, const int* in_sizes, int n_in,
                              void* d_out, int out_size, void* d_ws, size_t ws_size,
                              hipStream_t stream) {
    (void)in_sizes; (void)n_in; (void)out_size; (void)ws_size;

    const int*   z    = (const int*)  d_in[0];
    const float* pos  = (const float*)d_in[1];
    const float* bb   = (const float*)d_in[2];
    const int*   ei   = (const int*)  d_in[3];
    const int*   bidx = (const int*)  d_in[4];
    const float* Wemb = (const float*)d_in[5];
    const float* bemb = (const float*)d_in[6];
    const float* m1W  = (const float*)d_in[7];
    const float* m1b  = (const float*)d_in[8];
    const float* m2W0 = (const float*)d_in[9];
    const float* m2b0 = (const float*)d_in[10];
    const float* m2W  = (const float*)d_in[11];
    const float* m2b  = (const float*)d_in[12];
    const float* sym0 = (const float*)d_in[13];
    const float* sym1 = (const float*)d_in[14];
    const float* lWs  = (const float*)d_in[15];
    const float* lWv  = (const float*)d_in[16];
    const float* o1W  = (const float*)d_in[17];
    const float* o1b  = (const float*)d_in[18];
    const float* o2W  = (const float*)d_in[19];
    const float* o2b  = (const float*)d_in[20];
    float* out = (float*)d_out;

    // workspace layout (~13.2 MB), 256B aligned slices
    char* base = (char*)d_ws;
    size_t off = 0;
    auto alloc = [&](size_t bytes) -> char* {
        char* p = base + off;
        off = (off + bytes + 255) & ~(size_t)255;
        return p;
    };
    float*  sN  = (float*)alloc((size_t)N_NODES * 16 * 4);
    float*  vN  = (float*)alloc((size_t)N_NODES * 48 * 4);
    float*  hsA = (float*)alloc((size_t)N_NODES * 16 * 4);
    float*  hvA = (float*)alloc((size_t)N_NODES * 48 * 4);
    float*  y1g = (float*)alloc((size_t)N_EDGES * 3 * 4);
    float*  fts = (float*)alloc((size_t)N_EDGES * 8 * 4);
    float*  g   = (float*)alloc((size_t)N_GRAPHS * 64 * 4);
    const int PK0 = 32 * 8 * 32 * 16;     // layer-0 packed elems (512 cols)
    const int PKL = 80 * 8 * 32 * 16;     // layer-1/2 packed elems (1280 cols)
    __bf16* w2p = (__bf16*)alloc((size_t)(PK0 + 2 * PKL) * 2);

    // init
    k_zero<<<(N_NODES * 48 + 255) / 256, 256, 0, stream>>>(vN, N_NODES * 48);
    k_embed<<<(N_NODES + 255) / 256, 256, 0, stream>>>(z, bb, Wemb, bemb, sN);
    k_edge_geom<<<(N_EDGES + 255) / 256, 256, 0, stream>>>(ei, pos, y1g, fts);

    // pack mlp2 weights to bf16 B-operand layout (all 3 layers)
    k_pack_w2<<<(PK0 + 255) / 256, 256, 0, stream>>>(m2W0, w2p, 512, PK0);
    k_pack_w2<<<(PKL + 255) / 256, 256, 0, stream>>>(m2W, w2p + PK0, 1280, PKL);
    k_pack_w2<<<(PKL + 255) / 256, 256, 0, stream>>>(m2W + (size_t)256 * 1280,
                                                     w2p + PK0 + PKL, 1280, PKL);

    const float A0F = 0.25f;                 // 1/sqrt(EMB)
    const float A0L = 0.17677669529663687f;  // 1/sqrt(2*EMB)
    const float A1L = 0.14433756729740643f;  // 1/sqrt(3*EMB)

    for (int l = 0; l < 3; ++l) {
        k_zero<<<(N_NODES * 16 + 255) / 256, 256, 0, stream>>>(hsA, N_NODES * 16);
        k_zero<<<(N_NODES * 48 + 255) / 256, 256, 0, stream>>>(hvA, N_NODES * 48);
        if (l == 0) {
            k_edge_msg<true><<<N_EDGES / 64, 128, 0, stream>>>(
                ei, ei + N_EDGES, y1g, fts, sN, vN,
                m1W, m1b, w2p, m2b0, hsA, hvA, A0F, 0.f);
        } else {
            k_edge_msg<false><<<N_EDGES / 64, 128, 0, stream>>>(
                ei, ei + N_EDGES, y1g, fts, sN, vN,
                m1W + (size_t)l * 8 * 256, m1b + (size_t)l * 256,
                w2p + PK0 + (size_t)(l - 1) * PKL, m2b + (size_t)(l - 1) * 1280,
                hsA, hvA, A0L, A1L);
        }
        k_node_update<<<(N_NODES + 255) / 256, 256, 0, stream>>>(
            hsA, hvA, sym0 + (size_t)l * 80, sym1 + (size_t)l * 64,
            lWs + (size_t)l * 256, lWv + (size_t)l * 256, sN, vN);
    }

    k_zero<<<(N_GRAPHS * 64 + 255) / 256, 256, 0, stream>>>(g, N_GRAPHS * 64);
    k_pool<<<(N_NODES + 255) / 256, 256, 0, stream>>>(sN, vN, bidx, g);
    k_readout<<<1, 256, 0, stream>>>(g, o1W, o1b, o2W, o2b, out);
}
